// EfficientLocalAttention_4793183502520
// MI455X (gfx1250) — matrix operand order repaired and checked by reading, not compiled, not run on hardware
//
#include <hip/hip_runtime.h>

#define C_DIM 384
#define NHEAD 12
#define S49   49

typedef __attribute__((ext_vector_type(16))) __bf16 v16bf;
typedef __attribute__((ext_vector_type(8)))  float  v8f;
typedef __attribute__((ext_vector_type(4)))  unsigned int vu4;

union Frag16 { v16bf bf; vu4 u[2]; };
union Pack2  { __bf16 h[2]; unsigned u; };

__device__ __forceinline__ v16bf ld_frag(const __bf16* p0, const __bf16* p1) {
  Frag16 f;
  f.u[0] = *(const vu4*)p0;
  f.u[1] = *(const vu4*)p1;
  return f.bf;
}

__device__ __forceinline__ v8f wmma_bf16(v16bf a, v16bf b, v8f c) {
  return __builtin_amdgcn_wmma_f32_16x16x32_bf16(false, a, false, b, (short)0, c,
                                                 false, false);
}

// ---------------- LDS layout (units: __bf16 elements) ----------------
// Regions are phase-disjoint; aliases are barrier-protected.
#define X_OFF   0         // 64 x 392 (staged x window, bf16)
#define X_STR   392
#define Q_OFF   25088     // 12 heads x (64 rows x 40)
#define QK_STR  40
#define QK_HEAD 2560
#define K_OFF   55808     // 12 heads x (64 rows x 40)
#define VT_OFF  86528     // 12 heads x (32 rows x 72)  (V transposed: [e][s])
#define VT_STR  72
#define VT_HEAD 2304
#define P_OFF   0         // alias X+Q: 12 heads x (64 x 72) softmax probs
#define P_STR   72
#define P_HEAD  4608
#define AO_OFF  55808     // alias K: 64 x 392 attention output (all heads)
#define LDS_ELEMS 114176
#define LDS_BYTES (LDS_ELEMS * 2)

// ---------------- weight fp32 -> bf16 conversion ----------------
#define WQKV_ELEMS (3 * C_DIM * C_DIM)   // 442368
#define WPROJ_ELEMS (C_DIM * C_DIM)      // 147456

__global__ void ela_cvt_weights(const float* __restrict__ wq,
                                const float* __restrict__ wp,
                                __bf16* __restrict__ wqb,
                                __bf16* __restrict__ wpb) {
  int i = blockIdx.x * blockDim.x + threadIdx.x;
  int total = WQKV_ELEMS + WPROJ_ELEMS;
  if (i < WQKV_ELEMS)
    wqb[i] = (__bf16)wq[i];
  else if (i < total)
    wpb[i - WQKV_ELEMS] = (__bf16)wp[i - WQKV_ELEMS];
}

// ---------------- fully fused window attention ----------------
// grid = 1024 windows, block = 384 threads (12 wave32, wave == head)
__global__ __launch_bounds__(384) void ela_fused(
    const float*  __restrict__ x,       // (16, 3136, 384) f32
    const __bf16* __restrict__ wqkv,    // (1152, 384) bf16
    const __bf16* __restrict__ wproj,   // (384, 384) bf16
    const float*  __restrict__ bias,    // (384,) f32
    float*        __restrict__ out)     // (16, 3136, 384) f32
{
  extern __shared__ __bf16 lds[];

  const int tid  = threadIdx.x;
  const int lane = tid & 31;
  const int wave = tid >> 5;            // == head
  const int lm   = lane & 15;
  const int half = lane >> 4;

  const int win = blockIdx.x;
  const int b   = win >> 6;
  const int wi  = win & 63;
  const int wh  = wi >> 3;
  const int ww  = wi & 7;

  // ---- Phase 0: stage x window into LDS as bf16 (rows 49..63 zero) ----
  for (int i = tid; i < 64 * C_DIM; i += 384) {
    int s = i / C_DIM;
    int c = i - s * C_DIM;
    float v = 0.0f;
    if (s < S49) {
      int hh = wh * 7 + s / 7;
      int wp = ww * 7 + s % 7;
      int n  = hh * 56 + wp;
      v = x[((size_t)b * 3136 + n) * C_DIM + c];
    }
    lds[X_OFF + s * X_STR + c] = (__bf16)v;
  }
  __syncthreads();

  // ---- Phase 1: QKV GEMM. Each wave: 2 passes of 3 N-tiles x 4 M-tiles ----
  for (int jj = 0; jj < 2; ++jj) {
    const int nt0 = wave * 6 + jj * 3;
    v8f acc[3][4] = {};
    for (int kt = 0; kt < 12; ++kt) {
      const int k0 = kt * 32;
      v16bf bfrag[3];
#pragma unroll
      for (int jn = 0; jn < 3; ++jn) {
        const __bf16* bp =
            wqkv + (size_t)((nt0 + jn) * 16 + lm) * C_DIM + k0 + half * 8;
        bfrag[jn] = ld_frag(bp, bp + 16);
      }
#pragma unroll
      for (int mt = 0; mt < 4; ++mt) {
        const __bf16* ap = lds + X_OFF + (mt * 16 + lm) * X_STR + k0 + half * 8;
        v16bf af = ld_frag(ap, ap + 16);
#pragma unroll
        for (int jn = 0; jn < 3; ++jn)
          acc[jn][mt] = wmma_bf16(af, bfrag[jn], acc[jn][mt]);
      }
    }
    // scatter to q / k / v^T LDS regions (each 16-col tile: one head, one of q/k/v)
#pragma unroll
    for (int jn = 0; jn < 3; ++jn) {
      const int d0  = (nt0 + jn) * 16;
      const int t   = d0 / C_DIM;
      const int rem = d0 - t * C_DIM;
      const int hh  = rem >> 5;
      const int e0  = rem & 31;
      if (t == 2) {
        // V transposed: consecutive rows are contiguous -> packed u32 stores
        unsigned* dstu = (unsigned*)(lds + VT_OFF + hh * VT_HEAD);
        const int rowoff = (e0 + lm) * VT_STR;
#pragma unroll
        for (int mt = 0; mt < 4; ++mt) {
          const int mbase = mt * 16 + 8 * half;   // even
#pragma unroll
          for (int q2 = 0; q2 < 4; ++q2) {
            Pack2 p;
            p.h[0] = (__bf16)acc[jn][mt][2 * q2];
            p.h[1] = (__bf16)acc[jn][mt][2 * q2 + 1];
            dstu[((rowoff + mbase) >> 1) + q2] = p.u;
          }
        }
      } else {
        __bf16* dst = lds + (t == 0 ? Q_OFF : K_OFF) + hh * QK_HEAD;
#pragma unroll
        for (int mt = 0; mt < 4; ++mt)
#pragma unroll
          for (int r = 0; r < 8; ++r) {
            const int m = mt * 16 + r + 8 * half;
            dst[m * QK_STR + e0 + lm] = (__bf16)acc[jn][mt][r];
          }
      }
    }
  }
  __syncthreads();

  // ---- Phase 2: per-head attention. scores = q k^T * scale ----
  const int head = wave;
  const __bf16* qh = lds + Q_OFF + head * QK_HEAD;
  const __bf16* kh = lds + K_OFF + head * QK_HEAD;

  v16bf qfrag[4];
#pragma unroll
  for (int mt = 0; mt < 4; ++mt) {
    const __bf16* ap = qh + (mt * 16 + lm) * QK_STR + half * 8;
    qfrag[mt] = ld_frag(ap, ap + 16);
  }

  v8f sc[4][4];
#pragma unroll
  for (int mt = 0; mt < 4; ++mt)
#pragma unroll
    for (int nt = 0; nt < 4; ++nt) sc[mt][nt] = (v8f){};

#pragma unroll
  for (int nt = 0; nt < 4; ++nt) {
    const __bf16* bp = kh + (nt * 16 + lm) * QK_STR + half * 8;
    v16bf bf = ld_frag(bp, bp + 16);
#pragma unroll
    for (int mt = 0; mt < 4; ++mt)
      sc[mt][nt] = wmma_bf16(qfrag[mt], bf, sc[mt][nt]);
  }

  // ---- Phase 3: softmax in registers (row spans 16 lanes of this half) ----
  const float scale = 0.17677669529663687f;  // 32^-0.5
#pragma unroll
  for (int mt = 0; mt < 4; ++mt) {
#pragma unroll
    for (int r = 0; r < 8; ++r) {
      float vals[4];
      float mx = -1e30f;
#pragma unroll
      for (int nt = 0; nt < 4; ++nt) {
        float v = sc[mt][nt][r] * scale;
        if (nt * 16 + lm >= S49) v = -1e30f;   // mask padded keys
        vals[nt] = v;
        mx = fmaxf(mx, v);
      }
#pragma unroll
      for (int off = 8; off >= 1; off >>= 1)
        mx = fmaxf(mx, __shfl_xor(mx, off, 16));
      float sum = 0.0f;
#pragma unroll
      for (int nt = 0; nt < 4; ++nt) {
        float e = __expf(vals[nt] - mx);
        vals[nt] = e;
        sum += e;
      }
#pragma unroll
      for (int off = 8; off >= 1; off >>= 1)
        sum += __shfl_xor(sum, off, 16);
      float inv = 1.0f / sum;
#pragma unroll
      for (int nt = 0; nt < 4; ++nt) sc[mt][nt][r] = vals[nt] * inv;
    }
  }

  __syncthreads();  // everyone done reading q/k/x before P overwrites them

  // write probabilities (bf16) to wave-private LDS region
  __bf16* ph = lds + P_OFF + head * P_HEAD;
#pragma unroll
  for (int mt = 0; mt < 4; ++mt)
#pragma unroll
    for (int nt = 0; nt < 4; ++nt)
#pragma unroll
      for (int r = 0; r < 8; ++r) {
        const int m = mt * 16 + r + 8 * half;
        ph[m * P_STR + nt * 16 + lm] = (__bf16)sc[mt][nt][r];
      }

  // ---- Phase 4: out = P @ V  (K = 64 over 2 steps, N = 32) ----
  const __bf16* vth = lds + VT_OFF + head * VT_HEAD;
  __bf16* ao = lds + AO_OFF;

  v16bf vfrag[2][2];  // [kt][n2]
#pragma unroll
  for (int kt = 0; kt < 2; ++kt)
#pragma unroll
    for (int n2 = 0; n2 < 2; ++n2) {
      const __bf16* bp = vth + (n2 * 16 + lm) * VT_STR + kt * 32 + half * 8;
      vfrag[kt][n2] = ld_frag(bp, bp + 16);
    }

#pragma unroll
  for (int mt = 0; mt < 4; ++mt) {
    v8f o0 = {};
    v8f o1 = {};
#pragma unroll
    for (int kt = 0; kt < 2; ++kt) {
      const __bf16* ap = ph + (mt * 16 + lm) * P_STR + kt * 32 + half * 8;
      v16bf af = ld_frag(ap, ap + 16);
      o0 = wmma_bf16(af, vfrag[kt][0], o0);
      o1 = wmma_bf16(af, vfrag[kt][1], o1);
    }
#pragma unroll
    for (int r = 0; r < 8; ++r) {
      const int m = mt * 16 + r + 8 * half;
      ao[m * X_STR + head * 32 + lm]      = (__bf16)o0[r];
      ao[m * X_STR + head * 32 + 16 + lm] = (__bf16)o1[r];
    }
  }
  __syncthreads();  // all heads' attention output staged

  // ---- Phase 5: proj GEMM + bias, scatter f32 to (B,N,C) ----
  {
    const int d_lo = head * 32 + lm;
    const int d_hi = d_lo + 16;
    v8f pacc[2][4] = {};
    for (int kt = 0; kt < 12; ++kt) {
      const int k0 = kt * 32;
      const __bf16* bp0 = wproj + (size_t)d_lo * C_DIM + k0 + half * 8;
      const __bf16* bp1 = wproj + (size_t)d_hi * C_DIM + k0 + half * 8;
      v16bf b0 = ld_frag(bp0, bp0 + 16);
      v16bf b1 = ld_frag(bp1, bp1 + 16);
#pragma unroll
      for (int mt = 0; mt < 4; ++mt) {
        const __bf16* ap = lds + AO_OFF + (mt * 16 + lm) * X_STR + k0 + half * 8;
        v16bf af = ld_frag(ap, ap + 16);
        pacc[0][mt] = wmma_bf16(af, b0, pacc[0][mt]);
        pacc[1][mt] = wmma_bf16(af, b1, pacc[1][mt]);
      }
    }
    const float bv0 = bias[d_lo];
    const float bv1 = bias[d_hi];
#pragma unroll
    for (int mt = 0; mt < 4; ++mt) {
#pragma unroll
      for (int r = 0; r < 8; ++r) {
        const int m = mt * 16 + r + 8 * half;
        if (m < S49) {
          const int hh = wh * 7 + m / 7;
          const int wp = ww * 7 + m % 7;
          const size_t base = ((size_t)b * 3136 + hh * 56 + wp) * C_DIM;
          out[base + d_lo] = pacc[0][mt][r] + bv0;
          out[base + d_hi] = pacc[1][mt][r] + bv1;
        }
      }
    }
  }
}

extern "C" void kernel_launch(void* const* d_in, const int* in_sizes, int n_in,
                              void* d_out, int out_size, void* d_ws, size_t ws_size,
                              hipStream_t stream) {
  const float* x     = (const float*)d_in[0];
  const float* wqkv  = (const float*)d_in[1];
  const float* wproj = (const float*)d_in[2];
  const float* bproj = (const float*)d_in[3];
  float* out = (float*)d_out;

  __bf16* wqkv_bf  = (__bf16*)d_ws;
  __bf16* wproj_bf = wqkv_bf + WQKV_ELEMS;

  {
    int total = WQKV_ELEMS + WPROJ_ELEMS;
    int blk = 256;
    int grd = (total + blk - 1) / blk;
    ela_cvt_weights<<<grd, blk, 0, stream>>>(wqkv, wproj, wqkv_bf, wproj_bf);
  }

  hipFuncSetAttribute((const void*)ela_fused,
                      hipFuncAttributeMaxDynamicSharedMemorySize, LDS_BYTES);
  ela_fused<<<1024, 384, LDS_BYTES, stream>>>(x, wqkv_bf, wproj_bf, bproj, out);
}
